// CrossSubgConv_30030411334423
// MI455X (gfx1250) — matrix operand order repaired and checked by reading, not compiled, not run on hardware
//
#include <hip/hip_runtime.h>

// Problem constants (from reference)
#define BB 16
#define NN 96
#define DD 256

typedef __attribute__((ext_vector_type(16))) __bf16 v16bf;
typedef __attribute__((ext_vector_type(8)))  float  v8f;

union FragAB { v16bf v; unsigned int u[8]; };

// round-to-nearest-even f32 -> bf16 bit pattern
static __device__ __forceinline__ unsigned short f2bf(float f) {
    unsigned u = __float_as_uint(f);
    unsigned r = u + 0x7fffu + ((u >> 16) & 1u);
    return (unsigned short)(r >> 16);
}

// LDS region pitches (ushort units; even => u32-aligned pairs; *17 mod 64 bank spread)
#define XP 34   // X tile      96 x 32  (row r, kk)
#define WP 34   // W tile^T   256 x 32  (col e, kk)
#define H1P 258 // h1         96 x 256  (row k, d)   row-major -> layer2 A-frags
#define H2P 98  // h2^T      256 x 96   (d, k)       -> final B-frags
#define AP 98   // A[b]       96 x 96   (i, k)       -> final A-frags

#define OFF_X  0
#define OFF_W  (OFF_X + NN * XP)          // 3264
#define OFF_H1 (OFF_W + DD * WP)          // 11968
#define OFF_H2 (OFF_H1 + NN * H1P)        // 36736
#define OFF_A  (OFF_H2 + DD * H2P)        // 61824
#define SMEM_TOT (OFF_A + NN * AP)        // 71232 ushorts = 142464 B

// Workspace layout (bytes) for the pre-converted bf16 operands
#define WS_WT1 0                          // 256*256*2 = 131072
#define WS_WT2 131072                     // 131072
#define WS_ABF 262144                     // 16*96*96*2 = 294912
#define WS_NEED (262144 + 294912)         // 557056

// A-fragment (16x32 bf16, MxK) per CDNA5 ISA layout.
static __device__ __forceinline__ void loadA(FragAB& fr, const unsigned short* base,
                                             int pitch, int row0, int kbase, int lane) {
    const int m = lane & 15, hi = lane >> 4;
    const unsigned short* p = base + (row0 + m) * pitch;
#pragma unroll
    for (int v = 0; v < 8; ++v) {
        int kk = kbase + 2 * v + (((v >> 2) & 1) << 3) + (hi << 3);
        fr.u[v] = *(const unsigned int*)(p + kk);
    }
}

// B-fragment (32x16 bf16, KxN) from column-major storage base[col][k].
static __device__ __forceinline__ void loadB(FragAB& fr, const unsigned short* base,
                                             int pitch, int col0, int kbase, int lane) {
    const int n = lane & 15, hi = lane >> 4;
    const unsigned short* p = base + (col0 + n) * pitch;
#pragma unroll
    for (int v = 0; v < 8; ++v) {
        int kk = kbase + 2 * v + (((v >> 2) & 1) << 3) + (hi << 3);
        fr.u[v] = *(const unsigned int*)(p + kk);
    }
}

static __device__ __forceinline__ v8f wmma_bf16(const FragAB& a, const FragAB& b, v8f c) {
    return __builtin_amdgcn_wmma_f32_16x16x32_bf16(false, a.v, false, b.v,
                                                   (short)0, c, false, false);
}

// ---- one-shot prep: W1,W2 -> bf16 transposed; A -> bf16 (into workspace) ----
__global__ void __launch_bounds__(256)
prep_kernel(const float* __restrict__ W1, const float* __restrict__ W2,
            const float* __restrict__ A,
            unsigned short* __restrict__ Wt1, unsigned short* __restrict__ Wt2,
            unsigned short* __restrict__ Abf) {
    const int idx = blockIdx.x * 256 + threadIdx.x;
    if (idx < DD * DD) {                 // Wt[e][k] = W[k][e]
        const int e = idx >> 8, k = idx & 255;
        Wt1[idx] = f2bf(W1[(size_t)k * DD + e]);
        Wt2[idx] = f2bf(W2[(size_t)k * DD + e]);
    }
    if (idx < BB * NN * NN) Abf[idx] = f2bf(A[idx]);
}

template <bool PREP>
__global__ void __launch_bounds__(256)
fused_mlp_mp_kernel(const float* __restrict__ Xg,
                    const unsigned char* __restrict__ maskg,
                    const float* __restrict__ Ag,
                    const float* __restrict__ W1, const float* __restrict__ b1,
                    const float* __restrict__ W2, const float* __restrict__ b2,
                    const unsigned short* __restrict__ Wt1bf,
                    const unsigned short* __restrict__ Wt2bf,
                    const unsigned short* __restrict__ Abf,
                    float* __restrict__ outg) {
    __shared__ __align__(16) unsigned short smem[SMEM_TOT];
    __shared__ float smask[NN];

    unsigned short* s_x  = smem + OFF_X;
    unsigned short* s_w  = smem + OFF_W;
    unsigned short* s_h1 = smem + OFF_H1;
    unsigned short* s_h2 = smem + OFF_H2;
    unsigned short* s_a  = smem + OFF_A;

    const int j = blockIdx.x;   // 0..95
    const int b = blockIdx.y;   // 0..15
    const int tid  = threadIdx.x;
    const int lane = tid & 31;
    const int wv   = tid >> 5;        // 8 waves
    const int n0   = wv * 32;         // each wave owns 2 column tiles [n0, n0+32)
    const int nl   = lane & 15;
    const int hi   = lane >> 4;

    // ---- stage A[b] (bf16) and mask column j ----
    if (PREP) {
        const unsigned short* Ab = Abf + (size_t)b * NN * NN;
        for (int i = tid; i < NN * NN / 2; i += 256) {  // u32 copies
            int r = i / (NN / 2), c2 = (i - r * (NN / 2)) * 2;
            *(unsigned int*)&s_a[r * AP + c2] = *(const unsigned int*)&Ab[r * NN + c2];
        }
    } else {
        const float* Ab = Ag + (size_t)b * NN * NN;
        for (int i = tid; i < NN * NN; i += 256) {
            int r = i / NN, k = i - r * NN;
            s_a[r * AP + k] = f2bf(Ab[i]);
        }
    }
    {
        const unsigned char* mb = maskg + (size_t)b * NN * NN;
        if (tid < NN) smask[tid] = mb[(size_t)tid * NN + j] ? 1.0f : 0.0f;
    }

    v8f acc[12];
#pragma unroll
    for (int t = 0; t < 12; ++t)
#pragma unroll
        for (int e = 0; e < 8; ++e) acc[t][e] = 0.0f;

    // Rows of this workgroup's MLP: X[b, r, j, :], r = 0..95
    const float* Xrow = Xg + (size_t)b * NN * NN * DD + (size_t)j * DD;

    // ================= Layer 1: h1 = relu(X W1 + b1) =================
    for (int k0 = 0; k0 < DD; k0 += 32) {
        __syncthreads();  // previous-iteration readers done before overwrite
        // X tile 96x32 -> bf16 LDS (coalesced), speculative prefetch of next slice
        for (int i = tid; i < NN * 32; i += 256) {
            int dd = i & 31, r = i >> 5;
            const float* src = &Xrow[(size_t)r * NN * DD + k0 + dd];
            s_x[r * XP + dd] = f2bf(*src);
            if (k0 + 32 < DD) __builtin_prefetch(src + 32, 0, 0);
        }
        // W1 k-slice 32x256 -> transposed bf16 LDS
        if (PREP) {
            for (int i = tid; i < 32 * DD / 2; i += 256) {
                int e = i >> 4, rq = (i & 15) * 2;
                *(unsigned int*)&s_w[e * WP + rq] =
                    *(const unsigned int*)&Wt1bf[(size_t)e * DD + k0 + rq];
            }
        } else {
            for (int i = tid; i < 32 * DD; i += 256) {
                int e = i & 255, r = i >> 8;
                s_w[e * WP + r] = f2bf(W1[(size_t)(k0 + r) * DD + e]);
            }
        }
        __syncthreads();
        FragAB bf0, bf1;
        loadB(bf0, s_w, WP, n0, 0, lane);
        loadB(bf1, s_w, WP, n0 + 16, 0, lane);
#pragma unroll
        for (int mt = 0; mt < 6; ++mt) {
            FragAB af;
            loadA(af, s_x, XP, mt * 16, 0, lane);
            acc[mt * 2 + 0] = wmma_bf16(af, bf0, acc[mt * 2 + 0]);
            acc[mt * 2 + 1] = wmma_bf16(af, bf1, acc[mt * 2 + 1]);
        }
    }
    // epilogue: bias + relu -> h1 (row-major, bf16)
#pragma unroll
    for (int t = 0; t < 2; ++t) {
        const int ng = n0 + t * 16 + nl;
        const float bias = b1[ng];
#pragma unroll
        for (int mt = 0; mt < 6; ++mt)
#pragma unroll
            for (int v = 0; v < 8; ++v) {
                float val = acc[mt * 2 + t][v] + bias;
                val = fmaxf(val, 0.0f);
                s_h1[(mt * 16 + v + hi * 8) * H1P + ng] = f2bf(val);
            }
    }

#pragma unroll
    for (int t = 0; t < 12; ++t)
#pragma unroll
        for (int e = 0; e < 8; ++e) acc[t][e] = 0.0f;

    // ================= Layer 2: h2 = mask * relu(h1 W2 + b2) =================
    for (int k0 = 0; k0 < DD; k0 += 32) {
        __syncthreads();
        if (PREP) {
            for (int i = tid; i < 32 * DD / 2; i += 256) {
                int e = i >> 4, rq = (i & 15) * 2;
                *(unsigned int*)&s_w[e * WP + rq] =
                    *(const unsigned int*)&Wt2bf[(size_t)e * DD + k0 + rq];
            }
        } else {
            for (int i = tid; i < 32 * DD; i += 256) {
                int e = i & 255, r = i >> 8;
                s_w[e * WP + r] = f2bf(W2[(size_t)(k0 + r) * DD + e]);
            }
        }
        __syncthreads();
        FragAB bf0, bf1;
        loadB(bf0, s_w, WP, n0, 0, lane);
        loadB(bf1, s_w, WP, n0 + 16, 0, lane);
#pragma unroll
        for (int mt = 0; mt < 6; ++mt) {
            FragAB af;
            loadA(af, s_h1, H1P, mt * 16, k0, lane);
            acc[mt * 2 + 0] = wmma_bf16(af, bf0, acc[mt * 2 + 0]);
            acc[mt * 2 + 1] = wmma_bf16(af, bf1, acc[mt * 2 + 1]);
        }
    }
    // epilogue: bias + relu + mask -> h2 transposed [d][k]
#pragma unroll
    for (int t = 0; t < 2; ++t) {
        const int ng = n0 + t * 16 + nl;
        const float bias = b2[ng];
#pragma unroll
        for (int mt = 0; mt < 6; ++mt)
#pragma unroll
            for (int v = 0; v < 8; ++v) {
                const int kr = mt * 16 + v + hi * 8;   // row index = k
                float val = acc[mt * 2 + t][v] + bias;
                val = fmaxf(val, 0.0f) * smask[kr];
                s_h2[ng * H2P + kr] = f2bf(val);
            }
    }

#pragma unroll
    for (int t = 0; t < 12; ++t)
#pragma unroll
        for (int e = 0; e < 8; ++e) acc[t][e] = 0.0f;

    __syncthreads();  // h2t complete before anyone reads it

    // ================= Final: out[i,d] = sum_k A[b,i,k] * h2[k,d] =================
#pragma unroll
    for (int k0 = 0; k0 < NN; k0 += 32) {
        FragAB bf0, bf1;
        loadB(bf0, s_h2, H2P, n0, k0, lane);
        loadB(bf1, s_h2, H2P, n0 + 16, k0, lane);
#pragma unroll
        for (int mt = 0; mt < 6; ++mt) {
            FragAB af;
            loadA(af, s_a, AP, mt * 16, k0, lane);
            acc[mt * 2 + 0] = wmma_bf16(af, bf0, acc[mt * 2 + 0]);
            acc[mt * 2 + 1] = wmma_bf16(af, bf1, acc[mt * 2 + 1]);
        }
    }
    // write out[b, i, j, d] (f32)
    float* outp = outg + (size_t)b * NN * NN * DD + (size_t)j * DD;
#pragma unroll
    for (int t = 0; t < 2; ++t) {
        const int dcol = n0 + t * 16 + nl;
#pragma unroll
        for (int mt = 0; mt < 6; ++mt)
#pragma unroll
            for (int v = 0; v < 8; ++v) {
                const int irow = mt * 16 + v + hi * 8;
                outp[(size_t)irow * NN * DD + dcol] = acc[mt * 2 + t][v];
            }
    }
}

extern "C" void kernel_launch(void* const* d_in, const int* in_sizes, int n_in,
                              void* d_out, int out_size, void* d_ws, size_t ws_size,
                              hipStream_t stream) {
    (void)in_sizes; (void)n_in; (void)out_size;
    const float*         X    = (const float*)d_in[0];
    const unsigned char* mask = (const unsigned char*)d_in[1];
    const float*         A    = (const float*)d_in[2];
    const float*         W1   = (const float*)d_in[3];
    const float*         b1   = (const float*)d_in[4];
    const float*         W2   = (const float*)d_in[5];
    const float*         b2   = (const float*)d_in[6];
    float*               out  = (float*)d_out;

    dim3 grid(NN, BB);  // one workgroup per (j, b): 1536 workgroups

    if (d_ws != nullptr && ws_size >= (size_t)WS_NEED) {
        unsigned short* Wt1 = (unsigned short*)((char*)d_ws + WS_WT1);
        unsigned short* Wt2 = (unsigned short*)((char*)d_ws + WS_WT2);
        unsigned short* Abf = (unsigned short*)((char*)d_ws + WS_ABF);
        prep_kernel<<<(BB * NN * NN + 255) / 256, 256, 0, stream>>>(W1, W2, A, Wt1, Wt2, Abf);
        fused_mlp_mp_kernel<true><<<grid, 256, 0, stream>>>(
            X, mask, A, W1, b1, W2, b2, Wt1, Wt2, Abf, out);
    } else {
        fused_mlp_mp_kernel<false><<<grid, 256, 0, stream>>>(
            X, mask, A, W1, b1, W2, b2, nullptr, nullptr, nullptr, out);
    }
}
